// GMPool_masking_FC_67181878444619
// MI455X (gfx1250) — matrix operand order repaired and checked by reading, not compiled, not run on hardware
//
#include <hip/hip_runtime.h>
#include <hip/hip_bf16.h>
#include <math.h>

typedef __attribute__((ext_vector_type(16))) _Float16 v16h;
typedef __attribute__((ext_vector_type(8)))  _Float16 v8h;
typedef __attribute__((ext_vector_type(8)))  float    v8f;

#define B_    32
#define T_    150
#define N_    116
#define H_    64
#define TT    (T_*T_)          // 22500
#define R_    (B_*TT)          // 720000
#define NEIG  10
#define KSEL  105              // int(150*0.7)
#define NPAIR 6670             // 116*115/2
#define NSWEEP 12

__device__ inline v16h concat8(v8h lo, v8h hi) {
  return __builtin_shufflevector(lo, hi, 0,1,2,3,4,5,6,7,8,9,10,11,12,13,14,15);
}

// raw hardware sqrt/rsq (results feed f16 matrix inputs; no IEEE fixup needed)
__device__ inline float fast_sqrt(float x) { return __builtin_amdgcn_sqrtf(x); }
__device__ inline float fast_rsq(float x)  { return __builtin_amdgcn_rsqf(x); }

// ---------------------------------------------------------------------------
// A-fragments built straight from X (no LDS): lane holds row = lane&15;
// lane-half selects disjoint K ranges, so X rows are read exactly once/tile.
// A frag (16x32 f16): element e<8 -> K = kb+e, e>=8 -> K = kb+16+(e-8),
// kb = ks*32 + (lane>=16 ? 8 : 0).
// ---------------------------------------------------------------------------
__device__ inline void build_a_frags(const float* __restrict__ X, int r0,
                                     int lane, v16h afrag[4]) {
  const int rsel = lane & 15, half = lane >> 4;
  const int r = r0 + rsel;
  const int b = r / TT;
  const int rem = r - b * TT;
  const int t1 = rem / T_;
  const int t2 = rem - t1 * T_;
  const float* x1 = X + (b * T_ + t1) * N_;
  const float* x2 = X + (b * T_ + t2) * N_;
  __builtin_prefetch(x1, 0, 3);        // global_prefetch_b8
  __builtin_prefetch(x2, 0, 3);
#pragma unroll
  for (int ks = 0; ks < 4; ++ks) {
    const int kb = ks * 32 + half * 8;
    v16h a;
#pragma unroll
    for (int blk = 0; blk < 2; ++blk) {     // K runs kb.. and kb+16..
      const int kstart = kb + blk * 16;
#pragma unroll
      for (int u = 0; u < 2; ++u) {         // two aligned float4 chunks
        const int base = kstart + u * 4;
        float d0 = 0.f, d1 = 0.f, d2 = 0.f, d3 = 0.f;
        if (base < N_) {                    // base multiple of 4; 116 = 4*29
          const float4 pa = *(const float4*)(x1 + base);
          const float4 pb = *(const float4*)(x2 + base);
          const float e0 = pa.x - pb.x, e1 = pa.y - pb.y;
          const float e2 = pa.z - pb.z, e3 = pa.w - pb.w;
          d0 = fast_sqrt(e0 * e0 + 1e-9f); d1 = fast_sqrt(e1 * e1 + 1e-9f);
          d2 = fast_sqrt(e2 * e2 + 1e-9f); d3 = fast_sqrt(e3 * e3 + 1e-9f);
        }
        const int eb = blk * 8 + u * 4;
        a[eb + 0] = (_Float16)d0; a[eb + 1] = (_Float16)d1;
        a[eb + 2] = (_Float16)d2; a[eb + 3] = (_Float16)d3;
      }
    }
    afrag[ks] = a;
  }
}

// W1 staged fragment-major: sWf[kgroup][n] is one ready-to-load v16h B-frag.
// kgroup = ks*2+half ; element e<8 -> K = ks*32 + half*8 + e, e>=8 -> K = ks*32 + 16 + half*8 + (e-8).
__device__ inline void stage_w1(v16h (*sWf)[H_], const float* __restrict__ W1,
                                int tid) {
  _Float16* wf = (_Float16*)sWf;
  for (int idx = tid; idx < 8 * H_ * 16; idx += 256) {
    const int e = idx & 15;
    const int n = (idx >> 4) & 63;
    const int kg = idx >> 10;
    const int k = (kg >> 1) * 32 + (kg & 1) * 8 + e + ((e >= 8) ? 8 : 0);
    wf[idx] = (_Float16)((k < N_) ? W1[k * H_ + n] : 0.f);
  }
}

__device__ inline void compute_tile(const v16h afrag[4],
                                    const v16h (*sWf)[H_],
                                    v8f c[4], int lane) {
  const int half = lane >> 4;
  const int rsel = lane & 15;
#pragma unroll
  for (int nt = 0; nt < 4; ++nt) {
    v8f acc = {};
#pragma unroll
    for (int ks = 0; ks < 4; ++ks) {
      const v16h bfr = sWf[(ks << 1) | half][nt * 16 + rsel];  // 32B LDS load
      acc = __builtin_amdgcn_wmma_f32_16x16x32_f16(
          false, afrag[ks], false, bfr, (short)0, acc, false, false);
    }
    c[nt] = acc;
  }
}

// ---------------------------------------------------------------------------
__global__ void __launch_bounds__(256) k_init(float* stats) {
  if (threadIdx.x < 128) stats[threadIdx.x] = 0.f;
}

// Pass 1: WMMA GEMM + per-column sum / sumsq of h = dist@W1 + b1
__global__ void __launch_bounds__(256)
k_gemm_stats(const float* __restrict__ X, const float* __restrict__ W1,
             const float* __restrict__ b1, float* gSum, float* gSq) {
  __shared__ v16h sWf[8][H_];
  __shared__ float sSum[H_], sSq[H_];
  const int tid = threadIdx.x;
  const int w = tid >> 5, lane = tid & 31;
  stage_w1(sWf, W1, tid);
  if (tid < H_) { sSum[tid] = 0.f; sSq[tid] = 0.f; }
  const int r0 = (blockIdx.x * 8 + w) * 16;
  v16h afrag[4];
  build_a_frags(X, r0, lane, afrag);
  __syncthreads();

  v8f c[4];
  compute_tile(afrag, sWf, c, lane);

  const int rsel = lane & 15;
#pragma unroll
  for (int nt = 0; nt < 4; ++nt) {
    const int n = nt * 16 + rsel;
    const float bb = b1[n];
    float s = 0.f, q = 0.f;
#pragma unroll
    for (int vr = 0; vr < 8; ++vr) {
      const float h = c[nt][vr] + bb;
      s += h; q += h * h;
    }
    atomicAdd(&sSum[n], s);
    atomicAdd(&sSq[n], q);
  }
  __syncthreads();
  if (tid < H_) { atomicAdd(&gSum[tid], sSum[tid]); atomicAdd(&gSq[tid], sSq[tid]); }
}

__global__ void __launch_bounds__(64)
k_bn(const float* gSum, const float* gSq, const float* gamma,
     const float* beta, float* scale, float* shift) {
  const int n = threadIdx.x;
  if (n < H_) {
    const float inv  = 1.f / (float)R_;
    const float mean = gSum[n] * inv;
    const float var  = gSq[n] * inv - mean * mean;
    const float sc   = gamma[n] * fast_rsq(var + 1e-5f);
    scale[n] = sc;
    shift[n] = beta[n] - mean * sc;
  }
}

// Pass 2: recompute h via WMMA, BN+ReLU, dot with W2, sigmoid, diag=1 -> gm
__global__ void __launch_bounds__(256)
k_gemm_apply(const float* __restrict__ X, const float* __restrict__ W1,
             const float* __restrict__ b1, const float* __restrict__ W2,
             const float* __restrict__ b2, const float* __restrict__ scale,
             const float* __restrict__ shift, float* __restrict__ gm) {
  __shared__ v16h sWf[8][H_];
  const int tid = threadIdx.x;
  const int w = tid >> 5, lane = tid & 31;
  stage_w1(sWf, W1, tid);
  const int r0 = (blockIdx.x * 8 + w) * 16;
  v16h afrag[4];
  build_a_frags(X, r0, lane, afrag);
  __syncthreads();

  v8f c[4];
  compute_tile(afrag, sWf, c, lane);

  const int rsel = lane & 15, half = lane >> 4;
  float p[8];
#pragma unroll
  for (int vr = 0; vr < 8; ++vr) p[vr] = 0.f;
#pragma unroll
  for (int nt = 0; nt < 4; ++nt) {
    const int n = nt * 16 + rsel;
    const float bb = b1[n], sc = scale[n], sh = shift[n], w2 = W2[n];
#pragma unroll
    for (int vr = 0; vr < 8; ++vr) {
      float y = (c[nt][vr] + bb) * sc + sh;
      y = fmaxf(y, 0.f);
      p[vr] += y * w2;
    }
  }
  // reduce the 16 column-lanes within each half-wave
#pragma unroll
  for (int off = 1; off < 16; off <<= 1) {
#pragma unroll
    for (int vr = 0; vr < 8; ++vr) p[vr] += __shfl_xor(p[vr], off, 16);
  }
  if (rsel == 0) {
    const float bb2 = b2[0];
#pragma unroll
    for (int vr = 0; vr < 8; ++vr) {
      const int m = half * 8 + vr;
      const int r = r0 + m;
      const int b = r / TT;
      const int rem = r - b * TT;
      const int t1 = rem / T_;
      const int t2 = rem - t1 * T_;
      const float z = p[vr] + bb2;
      gm[r] = (t1 == t2) ? 1.f : 1.f / (1.f + __expf(-z));
    }
  }
}

// Parallel cyclic Jacobi eigensolver; A and V live in LDS (CDNA5: 320KB/WG)
__global__ void __launch_bounds__(256)
k_jacobi(const float* __restrict__ gm, float* __restrict__ vt) {
  extern __shared__ float sm[];
  float* Am = sm;                 // 22500
  float* Vm = Am + TT;            // 22500
  float* cc = Vm + TT;            // 75
  float* ss = cc + 75;            // 75
  int*   pp = (int*)(ss + 75);    // 75
  int*   qq = pp + 75;            // 75
  int*   sel = qq + 75;           // 16
  float* ev  = (float*)(sel + 16);// 150
  const int tid = threadIdx.x;
  const int b = blockIdx.x;
  const float* G = gm + b * TT;
  for (int i = tid; i < TT; i += 256) {
    Am[i] = G[i];
    const int rr = i / T_, ccol = i - rr * T_;
    Vm[i] = (rr == ccol) ? 1.f : 0.f;
  }
  __syncthreads();
  for (int sweep = 0; sweep < NSWEEP; ++sweep) {
    for (int round = 0; round < T_ - 1; ++round) {
      if (tid < 75) {
        int p, q;
        if (tid == 0) { p = T_ - 1; q = round; }
        else {
          p = (round + tid) % (T_ - 1);
          q = (round - tid + (T_ - 1)) % (T_ - 1);
        }
        if (p > q) { int t = p; p = q; q = t; }
        const float app = Am[p * T_ + p];
        const float aqq = Am[q * T_ + q];
        const float apq = Am[p * T_ + q];
        float c = 1.f, s = 0.f;
        if (fabsf(apq) > 1e-12f) {
          const float tau = (aqq - app) / (2.f * apq);
          const float t = (tau >= 0.f ? 1.f : -1.f) /
                          (fabsf(tau) + fast_sqrt(1.f + tau * tau));
          c = fast_rsq(1.f + t * t);
          s = t * c;
        }
        pp[tid] = p; qq[tid] = q; cc[tid] = c; ss[tid] = s;
      }
      __syncthreads();
      // column rotations (A and V); 75 disjoint pairs x 150 rows
      for (int idx = tid; idx < 75 * T_; idx += 256) {
        const int pr = idx / T_, row = idx - pr * T_;
        const int p = pp[pr], q = qq[pr];
        const float c = cc[pr], s = ss[pr];
        const float akp = Am[row * T_ + p], akq = Am[row * T_ + q];
        Am[row * T_ + p] = c * akp - s * akq;
        Am[row * T_ + q] = s * akp + c * akq;
        const float vkp = Vm[row * T_ + p], vkq = Vm[row * T_ + q];
        Vm[row * T_ + p] = c * vkp - s * vkq;
        Vm[row * T_ + q] = s * vkp + c * vkq;
      }
      __syncthreads();
      // row rotations (A only)
      for (int idx = tid; idx < 75 * T_; idx += 256) {
        const int pr = idx / T_, col = idx - pr * T_;
        const int p = pp[pr], q = qq[pr];
        const float c = cc[pr], s = ss[pr];
        const float apk = Am[p * T_ + col], aqk = Am[q * T_ + col];
        Am[p * T_ + col] = c * apk - s * aqk;
        Am[q * T_ + col] = s * apk + c * aqk;
      }
      __syncthreads();
    }
  }
  // top-10 eigenvalue indices (descending)
  for (int i = tid; i < T_; i += 256) ev[i] = Am[i * T_ + i];
  __syncthreads();
  if (tid == 0) {
    for (int j = 0; j < NEIG; ++j) {
      int arg = 0; float best = -1e30f;
      for (int i = 0; i < T_; ++i)
        if (ev[i] > best) { best = ev[i]; arg = i; }
      sel[j] = arg; ev[arg] = -1e30f;
    }
  }
  __syncthreads();
  for (int idx = tid; idx < NEIG * T_; idx += 256) {
    const int j = idx / T_, t = idx - j * T_;
    vt[(b * NEIG + j) * T_ + t] = fabsf(Vm[t * T_ + sel[j]]);
  }
}

// top-k (k=105) threshold per (b,eigvec) row via bitonic sort, emit 0/1 mask
__global__ void __launch_bounds__(256)
k_mask(const float* __restrict__ vt, float* __restrict__ mask) {
  __shared__ float v[256];
  __shared__ float thr_s;
  const int tid = threadIdx.x;
  const float* row = vt + blockIdx.x * T_;
  v[tid] = (tid < T_) ? row[tid] : -1e30f;
  __syncthreads();
  for (int k2 = 2; k2 <= 256; k2 <<= 1) {
    for (int j = k2 >> 1; j > 0; j >>= 1) {
      const int ixj = tid ^ j;
      if (ixj > tid) {
        const bool dir = ((tid & k2) == 0);
        const float a = v[tid], bb = v[ixj];
        if ((a > bb) == dir) { v[tid] = bb; v[ixj] = a; }
      }
      __syncthreads();
    }
  }
  if (tid == 0) thr_s = v[256 - KSEL];   // 105th-largest
  __syncthreads();
  if (tid < T_) mask[blockIdx.x * T_ + tid] = (row[tid] >= thr_s) ? 1.f : 0.f;
}

// Batched corrcoef: Gram of centered masked data via WMMA; constants cancel.
__global__ void __launch_bounds__(256)
k_corr(const float* __restrict__ X, const float* __restrict__ mask,
       float* __restrict__ out) {
  extern __shared__ char smem[];
  _Float16* xm  = (_Float16*)smem;                 // [128][160] f16, rows 320B
  float*    Cm  = (float*)(smem + 128 * 160 * 2);  // [128][128]
  float*    mk  = Cm + 128 * 128;                  // 160
  float*    rst = mk + 160;                        // 128
  const int tid = threadIdx.x;
  const int bk = blockIdx.x;            // (b, eig) flat, 320 blocks
  const int b = bk / NEIG;
  for (int t = tid; t < 160; t += 256) mk[t] = (t < T_) ? mask[bk * T_ + t] : 0.f;
  __syncthreads();
  if (tid < 128) {
    const int n = tid;
    const float* Xb = X + b * T_ * N_;
    float sum = 0.f;
    if (n < N_)
      for (int t = 0; t < T_; ++t) sum += Xb[t * N_ + n] * mk[t];
    const float mean = sum * (1.f / (float)T_);
    v8h* dst = (v8h*)(xm + n * 160);
    for (int g = 0; g < 20; ++g) {        // wide 16B LDS stores
      v8h h8;
#pragma unroll
      for (int u = 0; u < 8; ++u) {
        const int t = g * 8 + u;
        float v = 0.f;
        if (n < N_ && t < T_) v = Xb[t * N_ + n] * mk[t] - mean;
        h8[u] = (_Float16)v;
      }
      dst[g] = h8;
    }
  }
  __syncthreads();

  const int w = tid >> 5, lane = tid & 31;
  const int half = lane >> 4, rsel = lane & 15;
  {
    const int i = w;                       // row tile per wave (8 waves, 8 tiles)
    const _Float16* arow = xm + (i * 16 + rsel) * 160;
    v16h afrag[5];
#pragma unroll
    for (int ks = 0; ks < 5; ++ks) {
      const int kb = ks * 32 + half * 16;
      const v8h lo = *(const v8h*)(arow + kb);        // 16B LDS loads
      const v8h hi = *(const v8h*)(arow + kb + 8);
      afrag[ks] = concat8(lo, hi);
    }
    const _Float16* brow = xm + 0;
    for (int j = 0; j < 8; ++j) {          // col tiles; B = xm^T from same LDS
      const _Float16* bp = brow + (j * 16 + rsel) * 160;
      v8f acc = {};
#pragma unroll
      for (int ks = 0; ks < 5; ++ks) {
        const v16h bfr = *(const v16h*)(bp + ks * 32 + half * 16); // 32B load
        acc = __builtin_amdgcn_wmma_f32_16x16x32_f16(
            false, afrag[ks], false, bfr, (short)0, acc, false, false);
      }
#pragma unroll
      for (int vr = 0; vr < 8; ++vr) {
        const int m = half * 8 + vr;
        Cm[(i * 16 + m) * 128 + j * 16 + rsel] = acc[vr];
      }
    }
  }
  __syncthreads();
  if (tid < N_) rst[tid] = fast_rsq(fmaxf(Cm[tid * 128 + tid], 1e-30f));
  __syncthreads();
  for (int p = tid; p < NPAIR; p += 256) {
    int i = 0, rem = p;
    while (rem >= (N_ - 1 - i)) { rem -= (N_ - 1 - i); ++i; }
    const int j = i + 1 + rem;
    float val = Cm[i * 128 + j] * rst[i] * rst[j];
    val = fminf(fmaxf(val, -1.f), 1.f);
    out[bk * NPAIR + p] = val;
  }
}

// ---------------------------------------------------------------------------
extern "C" void kernel_launch(void* const* d_in, const int* in_sizes, int n_in,
                              void* d_out, int out_size, void* d_ws, size_t ws_size,
                              hipStream_t stream) {
  const float* X     = (const float*)d_in[0];
  const float* W1    = (const float*)d_in[1];
  const float* b1    = (const float*)d_in[2];
  const float* gamma = (const float*)d_in[3];
  const float* beta  = (const float*)d_in[4];
  const float* W2    = (const float*)d_in[5];
  const float* b2    = (const float*)d_in[6];
  float* ws    = (float*)d_ws;
  float* gSum  = ws;                  // 64
  float* gSq   = ws + 64;             // 64
  float* scale = ws + 128;            // 64
  float* shift = ws + 192;            // 64
  float* gm    = ws + 256;            // 720000
  float* vt    = gm + R_;             // 48000
  float* mask  = vt + B_ * NEIG * T_; // 48000
  float* out   = (float*)d_out;

  const int gemmBlocks = R_ / (16 * 8);   // 5625
  const size_t jacobiLds = (size_t)(TT * 2 + 75 * 4 + 16 + 160) * 4 + 256;
  const size_t corrLds   = 128 * 160 * 2 + 128 * 128 * 4 + (160 + 128) * 4;

  k_init<<<1, 128, 0, stream>>>(gSum);
  k_gemm_stats<<<gemmBlocks, 256, 0, stream>>>(X, W1, b1, gSum, gSq);
  k_bn<<<1, 64, 0, stream>>>(gSum, gSq, gamma, beta, scale, shift);
  k_gemm_apply<<<gemmBlocks, 256, 0, stream>>>(X, W1, b1, W2, b2, scale, shift, gm);
  k_jacobi<<<B_, 256, jacobiLds, stream>>>(gm, vt);
  k_mask<<<B_ * NEIG, 256, 0, stream>>>(vt, mask);
  k_corr<<<B_ * NEIG, 256, corrLds, stream>>>(X, mask, out);
}
